// LongDocClassifier_1881195676175
// MI455X (gfx1250) — compile-verified, matrix operand order checked by reading, not run on hardware
//
#include <hip/hip_runtime.h>
#include <hip/hip_bf16.h>

typedef __bf16 bf16_t;
typedef __attribute__((ext_vector_type(16))) __bf16 v16bf;
typedef __attribute__((ext_vector_type(8)))  __bf16 bf16x8;
typedef __attribute__((ext_vector_type(8)))  float  v8f;
typedef __attribute__((ext_vector_type(4)))  unsigned int u32x4;
typedef __attribute__((ext_vector_type(8)))  unsigned int u32x8;

#define SB   4096   // sequence length S
#define DD   768    // model dim D
#define HH   12     // heads
#define DHH  64     // head dim
#define WW   256    // window W1
#define GG   16     // global tokens
#define MTOK 8192   // B*S

__device__ __forceinline__ bf16_t f2bf(float f) {
  union { float f; unsigned u; } v; v.f = f;
  unsigned r = v.u + 0x7FFFu + ((v.u >> 16) & 1u);
  unsigned short h = (unsigned short)(r >> 16);
  return __builtin_bit_cast(bf16_t, h);
}

__device__ __forceinline__ v16bf ldAB(const bf16_t* p) {
  bf16x8 lo = *(const bf16x8*)(p);
  bf16x8 hi = *(const bf16x8*)(p + 16);
  v16bf r;
#pragma unroll
  for (int i = 0; i < 8; i++) { r[i] = lo[i]; r[i + 8] = hi[i]; }
  return r;
}

__device__ __forceinline__ v8f wmma_bf(v16bf a, v16bf b, v8f c) {
  return __builtin_amdgcn_wmma_f32_16x16x32_bf16(false, a, false, b, (short)0, c, false, false);
}

// ---- Tensor Data Mover: DMA a 2-D bf16 tile (tileX fast-dim elems x tileY
// rows, row stride strideElems) from global into LDS at ldsOff. D# built per
// CDNA5 ISA 8.3/8.4 (group0: count/lds/global/type, group1: dsize/dims/stride).
__device__ __forceinline__ void tdm_load_tile_2d(const bf16_t* g, unsigned ldsOff,
                                                 unsigned tileX, unsigned tileY,
                                                 unsigned strideElems) {
  unsigned long long ga = (unsigned long long)(size_t)g;
  u32x4 g0;
  g0[0] = 1u;                                              // count=1, user D#
  g0[1] = ldsOff;                                          // lds_addr
  g0[2] = (unsigned)ga;                                    // global_addr[31:0]
  g0[3] = (unsigned)((ga >> 32) & 0x01FFFFFFu) | (2u << 30); // ga[56:32] | type=2
  u32x8 g1;
  g1[0] = 1u << 16;                                        // data_size=1 (2 bytes)
  g1[1] = (tileX & 0xFFFFu) << 16;                         // tensor_dim0[15:0]
  g1[2] = ((tileX >> 16) & 0xFFFFu) | ((tileY & 0xFFFFu) << 16); // dim0 hi | dim1 lo
  g1[3] = ((tileY >> 16) & 0xFFFFu) | ((tileX & 0xFFFFu) << 16); // dim1 hi | tile_dim0
  g1[4] = tileY & 0xFFFFu;                                 // tile_dim1 (tile_dim2=0)
  g1[5] = strideElems;                                     // tensor_dim0_stride lo
  g1[6] = 0u;
  g1[7] = 0u;
  asm volatile("tensor_load_to_lds %0, %1" :: "s"(g0), "s"(g1) : "memory");
}

__device__ __forceinline__ unsigned ldsAddrOf(const void* p) {
  return (unsigned)(size_t)p;   // low 32 bits of flat shared addr = LDS offset
}

// ---------------- weight transpose+convert: Wt[n*K+k] = bf16(W[k*N+n]) -----
__global__ void wtrans_kernel(const float* __restrict__ W, bf16_t* __restrict__ Wt,
                              int K, int N) {
  size_t tid = (size_t)blockIdx.x * 256 + threadIdx.x;
  if (tid >= (size_t)K * N) return;
  int k = (int)(tid % K);
  int n = (int)(tid / K);
  Wt[tid] = f2bf(W[(size_t)k * N + n]);
}

// ---------------- v transpose: vt[((b*H+h)*DH+d)*S + s] = v[(b*S+s)*D+h*DH+d]
__global__ void vtrans_kernel(const bf16_t* __restrict__ v, bf16_t* __restrict__ vt) {
  size_t tid = (size_t)blockIdx.x * 256 + threadIdx.x;
  if (tid >= (size_t)2 * HH * DHH * SB) return;
  int s = (int)(tid % SB);
  size_t r = tid / SB;
  int d = (int)(r % DHH); r /= DHH;
  int h = (int)(r % HH);
  int b = (int)(r / HH);
  vt[tid] = v[((size_t)(b * SB + s)) * DD + h * DHH + d];
}

// ---------------- embedding gather ----------------------------------------
__global__ void embed_kernel(const int* __restrict__ ids, const float* __restrict__ tok,
                             const float* __restrict__ pos, float* __restrict__ y) {
  int row = blockIdx.x;                 // 0..MTOK-1
  int s = row & (SB - 1);
  int id = ids[row];
  const float* tr = tok + (size_t)id * DD;
  const float* pr = pos + (size_t)s * DD;
  float* yr = y + (size_t)row * DD;
  for (int d = threadIdx.x; d < DD; d += blockDim.x) yr[d] = tr[d] + pr[d];
}

// ---------------- layernorm (row per block) -------------------------------
__global__ __launch_bounds__(256) void ln_kernel(const float* __restrict__ y,
                                                 const float* __restrict__ g,
                                                 const float* __restrict__ b,
                                                 float* __restrict__ xo,
                                                 bf16_t* __restrict__ xob) {
  __shared__ float red[256];
  int tid = threadIdx.x;
  const float* yr = y + (size_t)blockIdx.x * DD;
  float s = 0.f;
  for (int i = tid; i < DD; i += 256) s += yr[i];
  red[tid] = s; __syncthreads();
  for (int st = 128; st > 0; st >>= 1) { if (tid < st) red[tid] += red[tid + st]; __syncthreads(); }
  float mean = red[0] / DD; __syncthreads();
  float s2 = 0.f;
  for (int i = tid; i < DD; i += 256) { float d = yr[i] - mean; s2 += d * d; }
  red[tid] = s2; __syncthreads();
  for (int st = 128; st > 0; st >>= 1) { if (tid < st) red[tid] += red[tid + st]; __syncthreads(); }
  float rstd = rsqrtf(red[0] / DD + 1e-5f);
  float* xr = xo + (size_t)blockIdx.x * DD;
  bf16_t* xbr = xob + (size_t)blockIdx.x * DD;
  for (int i = tid; i < DD; i += 256) {
    float v = (yr[i] - mean) * rstd * g[i] + b[i];
    xr[i] = v; xbr[i] = f2bf(v);
  }
}

// ---------------- TDM double-buffered WMMA GEMM ----------------------------
// C[M,N] = act((A[M,K] * Bt[N,K]^T + bias) * alpha) (+resid). 128x128 tile.
__global__ __launch_bounds__(256) void gemm_tdm(
    const bf16_t* __restrict__ A, const bf16_t* __restrict__ Bt,
    const float* __restrict__ bias, const float* __restrict__ resid,
    float* __restrict__ Cf, bf16_t* __restrict__ Cb,
    int M, int N, int K, int act, float alpha) {
  __shared__ alignas(128) bf16_t As[2][128][32];
  __shared__ alignas(128) bf16_t Bs[2][128][32];
  int wave = threadIdx.x >> 5, lane = threadIdx.x & 31;
  int half = lane >> 4, l16 = lane & 15;
  int rw = (wave >> 1) * 32;            // wave row offset within tile
  int cw = (wave & 1) * 64;             // wave col offset within tile
  int rowBase = blockIdx.x * 128;
  int colBase = blockIdx.y * 128;
  int nsteps = K >> 5;
  bool issuer = (threadIdx.x < 32);     // wave 0 drives the TDM

  if (issuer) {
    tdm_load_tile_2d(A  + (size_t)rowBase * K, ldsAddrOf(&As[0][0][0]), 32, 128, K);
    tdm_load_tile_2d(Bt + (size_t)colBase * K, ldsAddrOf(&Bs[0][0][0]), 32, 128, K);
  }
  v8f acc[2][4] = {};
  for (int s = 0; s < nsteps; s++) {
    int buf = s & 1;
    if (issuer) {
      if (s + 1 < nsteps) {
        int k0 = (s + 1) << 5;
        tdm_load_tile_2d(A  + (size_t)rowBase * K + k0, ldsAddrOf(&As[buf ^ 1][0][0]), 32, 128, K);
        tdm_load_tile_2d(Bt + (size_t)colBase * K + k0, ldsAddrOf(&Bs[buf ^ 1][0][0]), 32, 128, K);
        __builtin_amdgcn_s_wait_tensorcnt(2);   // stage s done; s+1 in flight
      } else {
        __builtin_amdgcn_s_wait_tensorcnt(0);
      }
    }
    __syncthreads();
    v16bf a0 = ldAB(&As[buf][rw + l16][half * 8]);
    v16bf a1 = ldAB(&As[buf][rw + 16 + l16][half * 8]);
#pragma unroll
    for (int j = 0; j < 4; j++) {
      v16bf b = ldAB(&Bs[buf][cw + j * 16 + l16][half * 8]);
      acc[0][j] = wmma_bf(a0, b, acc[0][j]);
      acc[1][j] = wmma_bf(a1, b, acc[1][j]);
    }
    __syncthreads();
  }
#pragma unroll
  for (int t = 0; t < 2; t++)
#pragma unroll
    for (int j = 0; j < 4; j++) {
      int col = colBase + cw + j * 16 + l16;
      float bi = bias ? bias[col] : 0.f;
#pragma unroll
      for (int i = 0; i < 8; i++) {
        int r = rowBase + rw + t * 16 + i + half * 8;
        if (r < M) {
          float v = (acc[t][j][i] + bi) * alpha;
          if (act == 1) { float u = v; v = 0.5f * u * (1.f + tanhf(0.7978845608f * (u + 0.044715f * u * u * u))); }
          size_t off = (size_t)r * N + col;
          if (resid) v += resid[off];
          if (Cf) Cf[off] = v;
          if (Cb) Cb[off] = f2bf(v);
        }
      }
    }
}

// ---------------- legacy per-lane GEMM (used for tiny row-gathered qg) -----
__global__ __launch_bounds__(256) void gemm_bf16(
    const bf16_t* __restrict__ A, const bf16_t* __restrict__ Bt,
    const float* __restrict__ bias, const float* __restrict__ resid,
    float* __restrict__ Cf, bf16_t* __restrict__ Cb,
    int M, int N, int K, int act, float alpha, int rowmap) {
  int wave = threadIdx.x >> 5, lane = threadIdx.x & 31;
  int half = lane >> 4, l16 = lane & 15;
  int rowBase = blockIdx.x * 128 + wave * 16;
  if (rowBase >= M) return;
  int colBase = blockIdx.y * 64;
  int m = rowBase + l16; if (m >= M) m = M - 1;
  int ar = rowmap ? ((m >> 4) * SB + (m & 15)) : m;   // qg gather: row = b*S + g
  const bf16_t* ap = A + (size_t)ar * K + half * 8;
  v8f acc[4] = {};
  for (int k0 = 0; k0 < K; k0 += 32) {
    v16bf a = ldAB(ap + k0);
#pragma unroll
    for (int j = 0; j < 4; j++) {
      const bf16_t* bp = Bt + (size_t)(colBase + j * 16 + l16) * K + k0 + half * 8;
      acc[j] = wmma_bf(a, ldAB(bp), acc[j]);
    }
  }
#pragma unroll
  for (int j = 0; j < 4; j++) {
    int col = colBase + j * 16 + l16;
    float bi = bias ? bias[col] : 0.f;
#pragma unroll
    for (int i = 0; i < 8; i++) {
      int r = rowBase + i + half * 8;
      if (r < M) {
        float v = (acc[j][i] + bi) * alpha;
        if (act == 1) { float u = v; v = 0.5f * u * (1.f + tanhf(0.7978845608f * (u + 0.044715f * u * u * u))); }
        size_t off = (size_t)r * N + col;
        if (resid) v += resid[off];
        if (Cf) Cf[off] = v;
        if (Cb) Cb[off] = f2bf(v);
      }
    }
  }
}

// ------- local sliding-window attention: flash-style, TDM-staged K/V -------
__global__ __launch_bounds__(256) void local_attn(
    const bf16_t* __restrict__ q, const bf16_t* __restrict__ k,
    const bf16_t* __restrict__ vt, const int* __restrict__ am,
    bf16_t* __restrict__ out) {
  int idx = blockIdx.x;
  int n = idx & 15; idx >>= 4;
  int h = idx % HH; int b = idx / HH;
  int wave = threadIdx.x >> 5, lane = threadIdx.x & 31;
  int half = lane >> 4, l16 = lane & 15;
  __shared__ alignas(128) bf16_t Ks[2][32][64];   // [key][dh]
  __shared__ alignas(128) bf16_t Vs[2][64][32];   // [dh][key]
  __shared__ alignas(64)  bf16_t P[8][16][32];

  int qb0 = n * WW + wave * 32;
  v16bf aQ[2][2];
#pragma unroll
  for (int t = 0; t < 2; t++) {
    int qrow = qb0 + t * 16 + l16;
    const bf16_t* qp = q + ((size_t)(b * SB + qrow)) * DD + h * DHH + half * 8;
    aQ[t][0] = ldAB(qp);
    aQ[t][1] = ldAB(qp + 32);
  }
  v8f o[2][4] = {};
  float mrow[2][8], lrow[2][8];
#pragma unroll
  for (int t = 0; t < 2; t++)
#pragma unroll
    for (int i = 0; i < 8; i++) { mrow[t][i] = -1e30f; lrow[t][i] = 0.f; }

  const float NEGV = -1e9f;
  int klo = (n - 1) * WW, khi = (n + 2) * WW;
  int lo = klo < 0 ? 0 : klo;
  int hi = khi > SB ? SB : khi;
  int nt = 1 + ((hi - lo) >> 5);        // tile 0 = 16 global keys (ks=0)
  const bf16_t* kBase  = k  + ((size_t)b * SB) * DD + h * DHH;
  const bf16_t* vtBase = vt + ((size_t)(b * HH + h) * DHH) * SB;
  bool issuer = (threadIdx.x < 32);

  if (issuer) {
    tdm_load_tile_2d(kBase, ldsAddrOf(&Ks[0][0][0]), 64, 32, DD);      // ks=0
    tdm_load_tile_2d(vtBase, ldsAddrOf(&Vs[0][0][0]), 32, 64, SB);
  }
  for (int ti = 0; ti < nt; ti++) {
    int buf = ti & 1;
    int ks = (ti == 0) ? 0 : lo + ((ti - 1) << 5);
    bool g16 = (ti == 0);
    if (issuer) {
      if (ti + 1 < nt) {
        int ksn = lo + (ti << 5);
        tdm_load_tile_2d(kBase + (size_t)ksn * DD, ldsAddrOf(&Ks[buf ^ 1][0][0]), 64, 32, DD);
        tdm_load_tile_2d(vtBase + ksn, ldsAddrOf(&Vs[buf ^ 1][0][0]), 32, 64, SB);
        __builtin_amdgcn_s_wait_tensorcnt(2);
      } else {
        __builtin_amdgcn_s_wait_tensorcnt(0);
      }
    }
    __syncthreads();
    int kp[2]; int ok[2];
    v8f sc[2][2] = {};
#pragma unroll
    for (int cg = 0; cg < 2; cg++) {
      kp[cg] = ks + cg * 16 + l16;
      ok[cg] = (am[b * SB + kp[cg]] != 0);
      v16bf b0 = ldAB(&Ks[buf][cg * 16 + l16][half * 8]);
      v16bf b1 = ldAB(&Ks[buf][cg * 16 + l16][32 + half * 8]);
#pragma unroll
      for (int t = 0; t < 2; t++) {
        sc[t][cg] = wmma_bf(aQ[t][0], b0, sc[t][cg]);
        sc[t][cg] = wmma_bf(aQ[t][1], b1, sc[t][cg]);
      }
    }
#pragma unroll
    for (int t = 0; t < 2; t++) {
      float sv[2][8];
#pragma unroll
      for (int cg = 0; cg < 2; cg++)
#pragma unroll
        for (int i = 0; i < 8; i++) {
          int qrow = qb0 + t * 16 + i + half * 8;
          bool valid;
          if (g16) valid = ok[cg] && (kp[cg] < GG);
          else {
            int dist = kp[cg] - qrow; if (dist < 0) dist = -dist;
            valid = ok[cg] && (kp[cg] >= GG) && (dist <= WW);
          }
          sv[cg][i] = valid ? sc[t][cg][i] : NEGV;
        }
      float rm[8];
#pragma unroll
      for (int i = 0; i < 8; i++) rm[i] = fmaxf(sv[0][i], sv[1][i]);
      for (int d = 1; d < 16; d <<= 1)
#pragma unroll
        for (int i = 0; i < 8; i++) rm[i] = fmaxf(rm[i], __shfl_xor(rm[i], d, 32));
      float p0[8], p1[8], rs[8];
#pragma unroll
      for (int i = 0; i < 8; i++) {
        float mn = fmaxf(mrow[t][i], rm[i]);
        float corr = __expf(mrow[t][i] - mn);
        mrow[t][i] = mn;
        p0[i] = (sv[0][i] > -5e8f) ? __expf(sv[0][i] - mn) : 0.f;
        p1[i] = (sv[1][i] > -5e8f) ? __expf(sv[1][i] - mn) : 0.f;
        rs[i] = p0[i] + p1[i];
        lrow[t][i] = lrow[t][i] * corr;
#pragma unroll
        for (int j = 0; j < 4; j++) o[t][j][i] = o[t][j][i] * corr;
      }
      for (int d = 1; d < 16; d <<= 1)
#pragma unroll
        for (int i = 0; i < 8; i++) rs[i] += __shfl_xor(rs[i], d, 32);
#pragma unroll
      for (int i = 0; i < 8; i++) lrow[t][i] += rs[i];
#pragma unroll
      for (int i = 0; i < 8; i++) {
        P[wave][i + half * 8][l16]      = f2bf(p0[i]);
        P[wave][i + half * 8][16 + l16] = f2bf(p1[i]);
      }
      v16bf pA = ldAB(&P[wave][l16][half * 8]);
#pragma unroll
      for (int j = 0; j < 4; j++) {
        v16bf bv = ldAB(&Vs[buf][j * 16 + l16][half * 8]);
        o[t][j] = wmma_bf(pA, bv, o[t][j]);
      }
    }
    __syncthreads();
  }
#pragma unroll
  for (int t = 0; t < 2; t++)
#pragma unroll
    for (int j = 0; j < 4; j++)
#pragma unroll
      for (int i = 0; i < 8; i++) {
        int qrow = qb0 + t * 16 + i + half * 8;
        float den = lrow[t][i];
        float val = (den > 0.f) ? o[t][j][i] / den : 0.f;
        out[((size_t)(b * SB + qrow)) * DD + h * DHH + j * 16 + l16] = f2bf(val);
      }
}

// ------- global attention: 8 waves split the keys, LSE-merge partials ------
__global__ __launch_bounds__(256) void global_attn(
    const bf16_t* __restrict__ qg, const bf16_t* __restrict__ kg,
    const bf16_t* __restrict__ vgt, const int* __restrict__ am,
    bf16_t* __restrict__ out) {
  int h = blockIdx.x % HH, b = blockIdx.x / HH;
  int wave = threadIdx.x >> 5, lane = threadIdx.x & 31;
  int half = lane >> 4, l16 = lane & 15;
  __shared__ alignas(64) bf16_t P[8][16][32];
  __shared__ float Osh[8][16][64];
  __shared__ float Msh[8][16], Lsh[8][16];
  const bf16_t* qp = qg + ((size_t)(b * GG + l16)) * DD + h * DHH + half * 8;
  v16bf a0 = ldAB(qp), a1 = ldAB(qp + 32);
  v8f o[4] = {};
  float mrow[8], lrow[8];
#pragma unroll
  for (int i = 0; i < 8; i++) { mrow[i] = -1e30f; lrow[i] = 0.f; }
  const float NEGV = -1e9f;
  int ks0 = wave * (SB / 8);
  for (int ks = ks0; ks < ks0 + SB / 8; ks += 32) {
    int kp[2]; int ok[2];
    v8f sc[2] = {};
#pragma unroll
    for (int cg = 0; cg < 2; cg++) {
      kp[cg] = ks + cg * 16 + l16;
      ok[cg] = (am[b * SB + kp[cg]] != 0);
      const bf16_t* kptr = kg + ((size_t)(b * SB + kp[cg])) * DD + h * DHH + half * 8;
      sc[cg] = wmma_bf(a0, ldAB(kptr), sc[cg]);
      sc[cg] = wmma_bf(a1, ldAB(kptr + 32), sc[cg]);
    }
    float sv[2][8];
#pragma unroll
    for (int cg = 0; cg < 2; cg++)
#pragma unroll
      for (int i = 0; i < 8; i++) sv[cg][i] = ok[cg] ? sc[cg][i] : NEGV;
    float rm[8];
#pragma unroll
    for (int i = 0; i < 8; i++) rm[i] = fmaxf(sv[0][i], sv[1][i]);
    for (int d = 1; d < 16; d <<= 1)
#pragma unroll
      for (int i = 0; i < 8; i++) rm[i] = fmaxf(rm[i], __shfl_xor(rm[i], d, 32));
    float p0[8], p1[8], rs[8];
#pragma unroll
    for (int i = 0; i < 8; i++) {
      float mn = fmaxf(mrow[i], rm[i]);
      float corr = __expf(mrow[i] - mn);
      mrow[i] = mn;
      p0[i] = (sv[0][i] > -5e8f) ? __expf(sv[0][i] - mn) : 0.f;
      p1[i] = (sv[1][i] > -5e8f) ? __expf(sv[1][i] - mn) : 0.f;
      rs[i] = p0[i] + p1[i];
      lrow[i] = lrow[i] * corr;
#pragma unroll
      for (int j = 0; j < 4; j++) o[j][i] = o[j][i] * corr;
    }
    for (int d = 1; d < 16; d <<= 1)
#pragma unroll
      for (int i = 0; i < 8; i++) rs[i] += __shfl_xor(rs[i], d, 32);
#pragma unroll
    for (int i = 0; i < 8; i++) lrow[i] += rs[i];
#pragma unroll
    for (int i = 0; i < 8; i++) {
      P[wave][i + half * 8][l16]      = f2bf(p0[i]);
      P[wave][i + half * 8][16 + l16] = f2bf(p1[i]);
    }
    v16bf pA = ldAB(&P[wave][l16][half * 8]);
#pragma unroll
    for (int j = 0; j < 4; j++) {
      const bf16_t* vp = vgt + (((size_t)(b * HH + h) * DHH) + j * 16 + l16) * SB + ks + half * 8;
      o[j] = wmma_bf(pA, ldAB(vp), o[j]);
    }
  }
#pragma unroll
  for (int j = 0; j < 4; j++)
#pragma unroll
    for (int i = 0; i < 8; i++) Osh[wave][i + half * 8][j * 16 + l16] = o[j][i];
#pragma unroll
  for (int i = 0; i < 8; i++) { Msh[wave][i + half * 8] = mrow[i]; Lsh[wave][i + half * 8] = lrow[i]; }
  __syncthreads();
  for (int e = threadIdx.x; e < GG * DHH; e += 256) {
    int g = e >> 6, d = e & 63;
    float m = -1e30f;
#pragma unroll
    for (int w = 0; w < 8; w++) m = fmaxf(m, Msh[w][g]);
    float osum = 0.f, lsum = 0.f;
#pragma unroll
    for (int w = 0; w < 8; w++) {
      float c = __expf(Msh[w][g] - m);
      osum += Osh[w][g][d] * c;
      lsum += Lsh[w][g] * c;
    }
    float val = (lsum > 0.f) ? osum / lsum : 0.f;
    out[((size_t)(b * SB + g)) * DD + h * DHH + d] = f2bf(val);
  }
}

// ---------------- classifier head ------------------------------------------
__global__ __launch_bounds__(256) void cls_kernel(
    const float* __restrict__ x, const float* __restrict__ Wcls,
    const float* __restrict__ bcls, const float* __restrict__ Wout,
    const float* __restrict__ bout, float* __restrict__ out) {
  int b = blockIdx.x, tid = threadIdx.x;
  const float* x0 = x + (size_t)b * SB * DD;
  __shared__ float c[DD];
  for (int j = tid; j < DD; j += 256) {
    float s = bcls[j];
    for (int i = 0; i < DD; i++) s += x0[i] * Wcls[(size_t)i * DD + j];
    c[j] = tanhf(s);
  }
  __syncthreads();
  __shared__ float r0[256], r1[256];
  float a0 = 0.f, a1 = 0.f;
  for (int j = tid; j < DD; j += 256) { a0 += c[j] * Wout[j * 2]; a1 += c[j] * Wout[j * 2 + 1]; }
  r0[tid] = a0; r1[tid] = a1; __syncthreads();
  for (int st = 128; st > 0; st >>= 1) {
    if (tid < st) { r0[tid] += r0[tid + st]; r1[tid] += r1[tid + st]; }
    __syncthreads();
  }
  if (tid == 0) { out[b * 2 + 0] = r0[0] + bout[0]; out[b * 2 + 1] = r1[0] + bout[1]; }
}

extern "C" void kernel_launch(void* const* d_in, const int* in_sizes, int n_in,
                              void* d_out, int out_size, void* d_ws, size_t ws_size,
                              hipStream_t stream) {
  (void)in_sizes; (void)n_in; (void)out_size; (void)ws_size;
  const int*   ids  = (const int*)d_in[0];
  const int*   amsk = (const int*)d_in[1];
  const float* tok  = (const float*)d_in[3];
  const float* pos  = (const float*)d_in[4];
  const float* eg   = (const float*)d_in[5];
  const float* ebv  = (const float*)d_in[6];
  const float* Wq   = (const float*)d_in[7];
  const float* Wk   = (const float*)d_in[8];
  const float* Wv   = (const float*)d_in[9];
  const float* Wqg  = (const float*)d_in[10];
  const float* Wkg  = (const float*)d_in[11];
  const float* Wvg  = (const float*)d_in[12];
  const float* Wo   = (const float*)d_in[13];
  const float* bq   = (const float*)d_in[14];
  const float* bk   = (const float*)d_in[15];
  const float* bv   = (const float*)d_in[16];
  const float* bqg  = (const float*)d_in[17];
  const float* bkg  = (const float*)d_in[18];
  const float* bvg  = (const float*)d_in[19];
  const float* bo   = (const float*)d_in[20];
  const float* ln1g = (const float*)d_in[21];
  const float* ln1b = (const float*)d_in[22];
  const float* Wf1  = (const float*)d_in[23];
  const float* bf1  = (const float*)d_in[24];
  const float* Wf2  = (const float*)d_in[25];
  const float* bf2  = (const float*)d_in[26];
  const float* ln2g = (const float*)d_in[27];
  const float* ln2b = (const float*)d_in[28];
  const float* Wcls = (const float*)d_in[29];
  const float* bcls = (const float*)d_in[30];
  const float* Wout = (const float*)d_in[31];
  const float* boutp= (const float*)d_in[32];

  size_t off = 0;
  char* base = (char*)d_ws;
  auto take = [&](size_t bytes) -> char* {
    char* r = base + off;
    off += (bytes + 255) & ~(size_t)255;
    return r;
  };
  float*  x    = (float*)take((size_t)MTOK * DD * 4);
  float*  y    = (float*)take((size_t)MTOK * DD * 4);
  bf16_t* xb   = (bf16_t*)take((size_t)MTOK * DD * 2);
  bf16_t* region = (bf16_t*)take((size_t)5 * MTOK * DD * 2); // q,k,v,kg,vg / reused as h
  bf16_t* qb  = region;
  bf16_t* kb  = region + (size_t)1 * MTOK * DD;
  bf16_t* vb  = region + (size_t)2 * MTOK * DD;
  bf16_t* kgb = region + (size_t)3 * MTOK * DD;
  bf16_t* vgb = region + (size_t)4 * MTOK * DD;
  bf16_t* hb  = region;                                      // alias (dead after attention)
  bf16_t* qgb = (bf16_t*)take((size_t)2 * GG * DD * 2);
  bf16_t* vt  = (bf16_t*)take((size_t)MTOK * DD * 2);
  bf16_t* vgt = (bf16_t*)take((size_t)MTOK * DD * 2);
  bf16_t* ao  = (bf16_t*)take((size_t)MTOK * DD * 2);
  bf16_t* wT[2][9];
  for (int l = 0; l < 2; l++) {
    for (int i = 0; i < 7; i++) wT[l][i] = (bf16_t*)take((size_t)DD * DD * 2);
    wT[l][7] = (bf16_t*)take((size_t)DD * 3072 * 2);  // Wf1^T [3072,768]
    wT[l][8] = (bf16_t*)take((size_t)3072 * DD * 2);  // Wf2^T [768,3072]
  }

  auto launchT = [&](const float* W, bf16_t* Wt, int K, int N) {
    size_t tot = (size_t)K * N;
    wtrans_kernel<<<dim3((unsigned)((tot + 255) / 256)), 256, 0, stream>>>(W, Wt, K, N);
  };
  for (int l = 0; l < 2; l++) {
    size_t o2 = (size_t)l * DD * DD;
    launchT(Wq  + o2, wT[l][0], DD, DD);
    launchT(Wk  + o2, wT[l][1], DD, DD);
    launchT(Wv  + o2, wT[l][2], DD, DD);
    launchT(Wqg + o2, wT[l][3], DD, DD);
    launchT(Wkg + o2, wT[l][4], DD, DD);
    launchT(Wvg + o2, wT[l][5], DD, DD);
    launchT(Wo  + o2, wT[l][6], DD, DD);
    launchT(Wf1 + (size_t)l * DD * 3072, wT[l][7], DD, 3072);
    launchT(Wf2 + (size_t)l * 3072 * DD, wT[l][8], 3072, DD);
  }

  embed_kernel<<<MTOK, 256, 0, stream>>>(ids, tok, pos, y);
  ln_kernel<<<MTOK, 256, 0, stream>>>(y, eg, ebv, x, xb);

  dim3 gD(MTOK / 128, DD / 128);     // 64 x 6
  dim3 gF(MTOK / 128, 3072 / 128);   // 64 x 24
  size_t vtN = (size_t)2 * HH * DHH * SB;
  for (int l = 0; l < 2; l++) {
    gemm_tdm<<<gD, 256, 0, stream>>>(xb, wT[l][0], bq + l * DD, nullptr, nullptr, qb,
                                     MTOK, DD, DD, 0, 0.125f);
    gemm_tdm<<<gD, 256, 0, stream>>>(xb, wT[l][1], bk + l * DD, nullptr, nullptr, kb,
                                     MTOK, DD, DD, 0, 1.f);
    gemm_tdm<<<gD, 256, 0, stream>>>(xb, wT[l][2], bv + l * DD, nullptr, nullptr, vb,
                                     MTOK, DD, DD, 0, 1.f);
    gemm_tdm<<<gD, 256, 0, stream>>>(xb, wT[l][4], bkg + l * DD, nullptr, nullptr, kgb,
                                     MTOK, DD, DD, 0, 1.f);
    gemm_tdm<<<gD, 256, 0, stream>>>(xb, wT[l][5], bvg + l * DD, nullptr, nullptr, vgb,
                                     MTOK, DD, DD, 0, 1.f);
    gemm_bf16<<<dim3(1, DD / 64), 256, 0, stream>>>(xb, wT[l][3], bqg + l * DD, nullptr,
                                     nullptr, qgb, 2 * GG, DD, DD, 0, 0.125f, 1);
    vtrans_kernel<<<dim3((unsigned)((vtN + 255) / 256)), 256, 0, stream>>>(vb, vt);
    vtrans_kernel<<<dim3((unsigned)((vtN + 255) / 256)), 256, 0, stream>>>(vgb, vgt);
    local_attn<<<2 * HH * 16, 256, 0, stream>>>(qb, kb, vt, amsk, ao);
    global_attn<<<2 * HH, 256, 0, stream>>>(qgb, kgb, vgt, amsk, ao);
    gemm_tdm<<<gD, 256, 0, stream>>>(ao, wT[l][6], bo + l * DD, x, y, nullptr,
                                     MTOK, DD, DD, 0, 1.f);
    ln_kernel<<<MTOK, 256, 0, stream>>>(y, ln1g + l * DD, ln1b + l * DD, x, xb);
    gemm_tdm<<<gF, 256, 0, stream>>>(xb, wT[l][7], bf1 + (size_t)l * 3072, nullptr,
                                     nullptr, hb, MTOK, 3072, DD, 1, 1.f);
    gemm_tdm<<<gD, 256, 0, stream>>>(hb, wT[l][8], bf2 + l * DD, x, y, nullptr,
                                     MTOK, DD, 3072, 0, 1.f);
    ln_kernel<<<MTOK, 256, 0, stream>>>(y, ln2g + l * DD, ln2b + l * DD, x, xb);
  }
  cls_kernel<<<2, 256, 0, stream>>>(x, Wcls, bcls, Wout, boutp, (float*)d_out);
}
